// SampleNet_63436666961941
// MI455X (gfx1250) — matrix operand order repaired and checked by reading, not compiled, not run on hardware
//
#include <hip/hip_runtime.h>
#include <hip/hip_bf16.h>

// ---------------- problem constants ----------------
#define QCH   256      // quantization channels (emb rows/cols)
#define CCH   128      // condition channels
#define AH    192      // GRU-A hidden
#define BH    32       // GRU-B hidden
#define TT    4096     // timesteps
#define BATCH 16       // batch
#define KA    896      // CCH + 3*QCH   (GRU-A input width)
#define NA    576      // 3*AH
#define KB    320      // AH + CCH      (GRU-B input width)
#define NB    96       // 3*BH
#define NF    512      // 2*QCH         (fc out)

#define KTA   28       // KA/32
#define KTB   10       // KB/32
#define NTA   36       // NA/16
#define NTB   6        // NB/16
#define NTF   32       // NF/16
#define NFRAG_A (NTA*KTA)            // 1008
#define NFRAG_B (NTB*KTB)            // 60
#define NFRAG_F (NTF)                // 32
#define NFRAG_TOT (NFRAG_A + NFRAG_B + NFRAG_F)   // 1100

typedef __bf16 bf16_t;
typedef __attribute__((ext_vector_type(16))) __bf16 v16bf;
typedef __attribute__((ext_vector_type(8)))  __bf16 v8bf;
typedef __attribute__((ext_vector_type(8)))  float  v8f;

__device__ inline v8f wmma_bf16(v16bf a, v16bf b, v8f c) {
  // D = A(16x32 bf16) * B(32x16 bf16) + C(16x16 f32)
  return __builtin_amdgcn_wmma_f32_16x16x32_bf16(
      /*neg_a=*/false, a, /*neg_b=*/false, b,
      /*c_mod=*/(short)0, c, /*reuse_a=*/false, /*reuse_b=*/false);
}

__device__ inline v16bf cat8(v8bf a, v8bf b) {
  return __builtin_shufflevector(a, b, 0,1,2,3,4,5,6,7,8,9,10,11,12,13,14,15);
}

__device__ inline v8bf cv8(const float* s) {
  v8bf r;
  #pragma unroll
  for (int i = 0; i < 8; ++i) r[i] = (__bf16)s[i];
  return r;
}

__device__ inline float sigm(float x) { return 1.0f / (1.0f + __expf(-x)); }

// ============================================================
// Kernel 0: one-time weight convert + swizzle into per-lane
// WMMA B-fragment layout (bf16). Fragment (nt,kt): lane holds
// N = nt*16 + (lane&15), K = kt*32 + (lane>>4)*8 + {[0,8), [16,24)}.
// ============================================================
__global__ __launch_bounds__(256) void k_prep(
    const float* __restrict__ Wa, const float* __restrict__ Wb,
    const float* __restrict__ Wf,
    bf16_t* __restrict__ swA, bf16_t* __restrict__ swB, bf16_t* __restrict__ swF)
{
  int t = blockIdx.x * 256 + threadIdx.x;
  int fid = t >> 5, lane = t & 31;
  if (fid >= NFRAG_TOT) return;

  const float* W; bf16_t* dst; int K, rel, kt;
  if (fid < NFRAG_A)                { W = Wa; dst = swA; K = KA; rel = fid;            kt = rel % KTA; }
  else if (fid < NFRAG_A + NFRAG_B) { W = Wb; dst = swB; K = KB; rel = fid - NFRAG_A;  kt = rel % KTB; }
  else                              { W = Wf; dst = swF; K = BH; rel = fid - NFRAG_A - NFRAG_B; kt = 0; }
  int nt = (fid < NFRAG_A) ? rel / KTA : (fid < NFRAG_A + NFRAG_B ? rel / KTB : rel);

  int n = nt * 16 + (lane & 15);
  const float* src = W + (size_t)n * K + kt * 32 + (lane >> 4) * 8;
  v16bf frag;
  #pragma unroll
  for (int i = 0; i < 16; ++i) frag[i] = (__bf16)src[i < 8 ? i : i + 8];
  *(v16bf*)(dst + ((size_t)rel * 32 + lane) * 16) = frag;
}

// ============================================================
// Kernel 1: fused embedding-interp + GEMM
//   xWa = concat(f, pe, se, ee) @ W_ih_a^T   (65536 x 576, K=896)
// One 16-row M tile per workgroup; concat panel staged once in
// LDS as bf16; 6 waves each compute 16M x 96N.
// ============================================================
__global__ __launch_bounds__(192) void k_xwa(
    const float* __restrict__ f,  const float* __restrict__ p,
    const float* __restrict__ sp, const float* __restrict__ ep_,
    const float* __restrict__ emb, const bf16_t* __restrict__ WswA,
    float* __restrict__ xWa)
{
  __shared__ bf16_t xa[BATCH * KA];        // 28672 B bf16 A panel
  __shared__ float  sfr[3][BATCH];
  __shared__ int    slo[3][BATCH];

  const int tid  = threadIdx.x;
  const int lane = tid & 31;
  const int wave = tid >> 5;               // 0..5
  const int m    = lane & 15;
  const int hi   = lane >> 4;
  const int mt   = blockIdx.x;

  // per-row interp params for the 3 embedding lookups
  if (tid < 48) {
    int which = tid >> 4, mm = tid & 15;
    const float* src = which == 0 ? p : (which == 1 ? sp : ep_);
    float v = src[mt * 16 + mm];
    int lo = (int)floorf(v); lo = lo < 0 ? 0 : (lo > QCH - 2 ? QCH - 2 : lo);
    slo[which][mm] = lo; sfr[which][mm] = v - (float)lo;
  }
  __syncthreads();

  // stage concat panel [16 x 896] -> bf16 LDS (built once, reused 6x)
  for (int i = tid; i < BATCH * KA; i += 192) {
    int mm = i / KA, k = i - mm * KA;
    float v;
    if (k < CCH) {
      v = f[(size_t)(mt * 16 + mm) * CCH + k];
    } else {
      int e = k - CCH;
      int which = e >> 8, cc = e & (QCH - 1);
      const float* b0 = emb + (size_t)slo[which][mm] * QCH + cc;
      float a0 = b0[0], a1 = b0[QCH];
      v = a0 + (a1 - a0) * sfr[which][mm];
    }
    xa[i] = (bf16_t)v;
  }
  __syncthreads();

  v8f acc[6];
  #pragma unroll
  for (int j = 0; j < 6; ++j)
    #pragma unroll
    for (int g = 0; g < 8; ++g) acc[j][g] = 0.0f;

  for (int kt = 0; kt < KTA; ++kt) {
    const bf16_t* ar = xa + m * KA + kt * 32 + hi * 8;
    v16bf afr = cat8(*(const v8bf*)ar, *(const v8bf*)(ar + 16));
    #pragma unroll
    for (int j = 0; j < 6; ++j) {
      int ntG = wave * 6 + j;
      v16bf bfr = *(const v16bf*)(WswA + ((size_t)(ntG * KTA + kt) * 32 + lane) * 16);
      acc[j] = wmma_bf16(afr, bfr, acc[j]);
    }
  }

  #pragma unroll
  for (int j = 0; j < 6; ++j) {
    int n = wave * 96 + j * 16 + m;
    #pragma unroll
    for (int g = 0; g < 8; ++g)
      xWa[(size_t)(mt * 16 + g + 8 * hi) * NA + n] = acc[j][g];
  }
}

// ============================================================
// Kernel 2: GRU-A sequential scan (single persistent workgroup)
//   per step: hW = h[16,192] @ W_hh_a^T (36 n-tiles x 6 k-tiles)
// W_hh_a bf16 fragments + h (f32 + bf16 mirror) resident in LDS.
// block = 384 (12 waves x 3 n-tiles); dyn LDS = 276480 B
// ============================================================
__global__ __launch_bounds__(384) void k_scanA(
    const float* __restrict__ xWa, const float* __restrict__ Whh,
    bf16_t* __restrict__ hA)
{
  extern __shared__ char smem[];
  bf16_t* whh  = (bf16_t*)smem;                      // 36*6*32*16*2 = 221184 B
  float*  hlds = (float*)(smem + 221184);            // [16][192] f32 = 12288 B
  bf16_t* hbf  = (bf16_t*)(smem + 233472);           // [16][192] bf16 = 6144 B
  float*  hw   = (float*)(smem + 239616);            // [16][576] f32 = 36864 B

  const int tid  = threadIdx.x;
  const int lane = tid & 31;
  const int wave = tid >> 5;          // 0..11
  const int m    = lane & 15;
  const int hi   = lane >> 4;

  // preload W_hh_a as bf16 B-fragments
  for (int q = 0; q < 18; ++q) {
    int ntG = wave * 3 + q / 6;
    int kt  = q % 6;
    int n   = ntG * 16 + m;
    const float* src = Whh + (size_t)n * AH + kt * 32 + hi * 8;
    v16bf frag;
    #pragma unroll
    for (int i = 0; i < 16; ++i) frag[i] = (__bf16)src[i < 8 ? i : i + 8];
    *(v16bf*)(whh + ((size_t)(ntG * 6 + kt) * 32 + lane) * 16) = frag;
  }
  for (int i = tid; i < BATCH * AH; i += 384) { hlds[i] = 0.0f; hbf[i] = (bf16_t)0.0f; }
  __syncthreads();

  // gate-element assignment + one-step-ahead register prefetch of xWa
  int gb[8], gj[8];
  const float* gx[8];
  float px[8], pz[8], pq[8];
  #pragma unroll
  for (int u = 0; u < 8; ++u) {
    int idx = tid + u * 384;          // 3072 = 16*192 gate elems
    gb[u] = idx / AH;
    gj[u] = idx - gb[u] * AH;
    gx[u] = xWa + (size_t)gb[u] * TT * NA + gj[u];
    px[u] = gx[u][0]; pz[u] = gx[u][AH]; pq[u] = gx[u][2 * AH];
  }

  for (int t = 0; t < TT; ++t) {
    // ---- stage 1: hW = h @ W_hh^T (pure ds loads + WMMA) ----
    v8f acc[3];
    #pragma unroll
    for (int j = 0; j < 3; ++j)
      #pragma unroll
      for (int g = 0; g < 8; ++g) acc[j][g] = 0.0f;

    #pragma unroll
    for (int kt = 0; kt < 6; ++kt) {
      const bf16_t* ar = hbf + m * AH + kt * 32 + hi * 8;
      v16bf afr = cat8(*(const v8bf*)ar, *(const v8bf*)(ar + 16));
      #pragma unroll
      for (int j = 0; j < 3; ++j) {
        int ntG = wave * 3 + j;
        v16bf bfr = *(const v16bf*)(whh + ((size_t)(ntG * 6 + kt) * 32 + lane) * 16);
        acc[j] = wmma_bf16(afr, bfr, acc[j]);
      }
    }
    #pragma unroll
    for (int j = 0; j < 3; ++j) {
      int n = (wave * 3 + j) * 16 + m;
      #pragma unroll
      for (int g = 0; g < 8; ++g) hw[(g + 8 * hi) * NA + n] = acc[j][g];
    }
    __syncthreads();

    // ---- stage 2: fused GRU gates, update h (f32 + bf16), emit hA ----
    #pragma unroll
    for (int u = 0; u < 8; ++u) {
      int b = gb[u], j = gj[u];
      float hr = hw[b * NA + j];
      float hz = hw[b * NA + AH + j];
      float hn = hw[b * NA + 2 * AH + j];
      float ho = hlds[b * AH + j];
      float r  = sigm(px[u] + hr);
      float z  = sigm(pz[u] + hz);
      float ng = tanhf(pq[u] + r * hn);
      float hn2 = (1.0f - z) * ng + z * ho;
      hlds[b * AH + j] = hn2;
      bf16_t hb16 = (bf16_t)hn2;
      hbf[b * AH + j] = hb16;
      hA[(size_t)(b * TT + t) * AH + j] = hb16;
    }
    // prefetch next step's input projections (hide L2 latency under WMMA)
    if (t + 1 < TT) {
      #pragma unroll
      for (int u = 0; u < 8; ++u) {
        gx[u] += NA;
        px[u] = gx[u][0]; pz[u] = gx[u][AH]; pq[u] = gx[u][2 * AH];
      }
    }
    __syncthreads();
  }
}

// ============================================================
// Kernel 3: fused concat + GEMM  xWb = [hA | f] @ W_ih_b^T
//   K = 320 (192 bf16 + 128 f32 cvt), N = 96
// grid = 512, block = 256 (8 waves, wave = one 16-row M tile)
// ============================================================
__global__ __launch_bounds__(256) void k_xwb(
    const bf16_t* __restrict__ hA, const float* __restrict__ f,
    const bf16_t* __restrict__ WswB, float* __restrict__ xWb)
{
  const int lane = threadIdx.x & 31;
  const int wave = threadIdx.x >> 5;
  const int m    = lane & 15;
  const int hi   = lane >> 4;
  const int mt   = blockIdx.x * 8 + wave;
  const int row  = mt * 16 + m;

  const bf16_t* hrow = hA + (size_t)row * AH;
  const float*  frow = f + (size_t)row * CCH;

  v8f acc[6];
  #pragma unroll
  for (int j = 0; j < 6; ++j)
    #pragma unroll
    for (int g = 0; g < 8; ++g) acc[j][g] = 0.0f;

  #pragma unroll
  for (int kt = 0; kt < KTB; ++kt) {
    int s0 = kt * 32 + hi * 8;
    int s1 = s0 + 16;
    v8bf r0 = (s0 < AH) ? *(const v8bf*)(hrow + s0) : cv8(frow + (s0 - AH));
    v8bf r1 = (s1 < AH) ? *(const v8bf*)(hrow + s1) : cv8(frow + (s1 - AH));
    v16bf afr = cat8(r0, r1);
    #pragma unroll
    for (int j = 0; j < 6; ++j) {
      v16bf bfr = *(const v16bf*)(WswB + ((size_t)(j * KTB + kt) * 32 + lane) * 16);
      acc[j] = wmma_bf16(afr, bfr, acc[j]);
    }
  }
  #pragma unroll
  for (int j = 0; j < 6; ++j) {
    int n = j * 16 + m;
    #pragma unroll
    for (int g = 0; g < 8; ++g)
      xWb[(size_t)(mt * 16 + g + 8 * hi) * NB + n] = acc[j][g];
  }
}

// ============================================================
// Kernel 4: GRU-B sequential scan (single workgroup)
// block = 192 (6 waves, one 16x16 n-tile each); dyn LDS = 15360 B
// ============================================================
__global__ __launch_bounds__(192) void k_scanB(
    const float* __restrict__ xWb, const float* __restrict__ Whh,
    bf16_t* __restrict__ hB)
{
  extern __shared__ char smem[];
  bf16_t* whh  = (bf16_t*)smem;               // 6*32*16*2 = 6144 B
  float*  hb   = (float*)(smem + 6144);       // [16][32] f32 = 2048 B
  bf16_t* hbbf = (bf16_t*)(smem + 8192);      // [16][32] bf16 = 1024 B
  float*  hw   = (float*)(smem + 9216);       // [16][96] f32 = 6144 B

  const int tid  = threadIdx.x;
  const int lane = tid & 31;
  const int wave = tid >> 5;                  // 0..5 -> n-tile
  const int m    = lane & 15;
  const int hi   = lane >> 4;

  { // preload W_hh_b fragment
    int n = wave * 16 + m;
    const float* src = Whh + (size_t)n * BH + hi * 8;
    v16bf frag;
    #pragma unroll
    for (int i = 0; i < 16; ++i) frag[i] = (__bf16)src[i < 8 ? i : i + 8];
    *(v16bf*)(whh + ((size_t)wave * 32 + lane) * 16) = frag;
  }
  for (int i = tid; i < BATCH * BH; i += 192) { hb[i] = 0.0f; hbbf[i] = (bf16_t)0.0f; }
  __syncthreads();

  for (int t = 0; t < TT; ++t) {
    v8f acc;
    #pragma unroll
    for (int g = 0; g < 8; ++g) acc[g] = 0.0f;
    const bf16_t* ar = hbbf + m * BH + hi * 8;
    v16bf afr = cat8(*(const v8bf*)ar, *(const v8bf*)(ar + 16));
    v16bf bfr = *(const v16bf*)(whh + ((size_t)wave * 32 + lane) * 16);
    acc = wmma_bf16(afr, bfr, acc);
    #pragma unroll
    for (int g = 0; g < 8; ++g) hw[(g + 8 * hi) * NB + wave * 16 + m] = acc[g];
    __syncthreads();

    for (int idx = tid; idx < BATCH * BH; idx += 192) {
      int b = idx >> 5, j = idx & 31;
      size_t base = (size_t)(b * TT + t) * NB + j;
      float r  = sigm(xWb[base]           + hw[b * NB + j]);
      float z  = sigm(xWb[base + BH]      + hw[b * NB + BH + j]);
      float ng = tanhf(xWb[base + 2 * BH] + r * hw[b * NB + 2 * BH + j]);
      float hn2 = (1.0f - z) * ng + z * hb[b * BH + j];
      hb[b * BH + j] = hn2;
      bf16_t h16 = (bf16_t)hn2;
      hbbf[b * BH + j] = h16;
      hB[(size_t)(b * TT + t) * BH + j] = h16;
    }
    __syncthreads();
  }
}

// ============================================================
// Kernel 5: out = pairsum( tanh(hB @ fc_W^T + b) * a )
//   M = 65536, N = 512, K = 32 (single bf16 WMMA K-tile)
// grid = 2048, block = 256 (8 waves: 2 M-tiles x 4 N-slices)
// ============================================================
__global__ __launch_bounds__(256) void k_fc(
    const bf16_t* __restrict__ hB, const bf16_t* __restrict__ WswF,
    const float* __restrict__ fcb, const float* __restrict__ av,
    float* __restrict__ out)
{
  const int lane = threadIdx.x & 31;
  const int wave = threadIdx.x >> 5;
  const int m    = lane & 15;
  const int hi   = lane >> 4;
  const int mt   = blockIdx.x * 2 + (wave >> 2);
  const int n0   = (wave & 3) * 128;

  const bf16_t* hrow = hB + (size_t)(mt * 16 + m) * BH;
  v16bf afr = cat8(*(const v8bf*)(hrow + hi * 8), *(const v8bf*)(hrow + 16 + hi * 8));

  v8f acc[8];
  #pragma unroll
  for (int j = 0; j < 8; ++j) {
    #pragma unroll
    for (int g = 0; g < 8; ++g) acc[j][g] = 0.0f;
    int ntG = (wave & 3) * 8 + j;
    v16bf bfr = *(const v16bf*)(WswF + ((size_t)ntG * 32 + lane) * 16);
    acc[j] = wmma_bf16(afr, bfr, acc[j]);
  }

  #pragma unroll
  for (int j = 0; j < 8; ++j) {
    int n = n0 + j * 16 + m;
    float bn = fcb[n], an = av[n];
    #pragma unroll
    for (int g = 0; g < 8; ++g) {
      float v  = tanhf(acc[j][g] + bn) * an;
      float vp = v + __shfl_xor(v, 1, 32);     // pair (2q, 2q+1) = adjacent lanes
      if ((lane & 1) == 0)
        out[(size_t)(mt * 16 + g + 8 * hi) * QCH + (n >> 1)] = vp;
    }
  }
}

// ============================================================
extern "C" void kernel_launch(void* const* d_in, const int* in_sizes, int n_in,
                              void* d_out, int out_size, void* d_ws, size_t ws_size,
                              hipStream_t stream) {
  const float* f     = (const float*)d_in[0];
  const float* p     = (const float*)d_in[1];
  const float* s_prv = (const float*)d_in[2];
  const float* e_prv = (const float*)d_in[3];
  const float* emb   = (const float*)d_in[4];
  const float* Wih_a = (const float*)d_in[5];
  const float* Whh_a = (const float*)d_in[6];
  const float* Wih_b = (const float*)d_in[7];
  const float* Whh_b = (const float*)d_in[8];
  const float* fcW   = (const float*)d_in[9];
  const float* fcb   = (const float*)d_in[10];
  const float* av    = (const float*)d_in[11];
  float* out = (float*)d_out;

  char* ws = (char*)d_ws;
  float*  xWa  = (float*)ws;                           // 65536*576*4 = 150,994,944 B
  bf16_t* hA   = (bf16_t*)(ws + 150994944);            // 65536*192*2 =  25,165,824 B
  float*  xWb  = (float*)(ws + 176160768);             // 65536*96*4  =  25,165,824 B
  bf16_t* hB   = (bf16_t*)(ws + 201326592);            // 65536*32*2  =   4,194,304 B
  bf16_t* WswA = (bf16_t*)(ws + 205520896);            // 1008*512*2  =   1,032,192 B
  bf16_t* WswB = (bf16_t*)(ws + 206553088);            //   60*512*2  =      61,440 B
  bf16_t* WswF = (bf16_t*)(ws + 206614528);            //   32*512*2  =      32,768 B

  k_prep <<<(NFRAG_TOT * 32 + 255) / 256, 256, 0, stream>>>(Wih_a, Wih_b, fcW, WswA, WswB, WswF);
  k_xwa  <<<4096, 192, 0,      stream>>>(f, p, s_prv, e_prv, emb, WswA, xWa);
  k_scanA<<<1,    384, 276480, stream>>>(xWa, Whh_a, hA);
  k_xwb  <<<512,  256, 0,      stream>>>(hA, f, WswB, xWb);
  k_scanB<<<1,    192, 15360,  stream>>>(xWb, Whh_b, hB);
  k_fc   <<<2048, 256, 0,      stream>>>(hB, WswF, fcb, av, out);
}
